// LinearAttention_43800076484693
// MI455X (gfx1250) — matrix operand order repaired and checked by reading, not compiled, not run on hardware
//
#include <hip/hip_runtime.h>

// ---------------------------------------------------------------------------
// Types for CDNA5 WMMA (wave32, 16x16x32 bf16 -> f32)
// ---------------------------------------------------------------------------
typedef __attribute__((ext_vector_type(16))) __bf16 v16bf;
typedef __attribute__((ext_vector_type(8)))  __bf16 v8bf;
typedef __attribute__((ext_vector_type(8)))  float  v8f;
typedef __attribute__((ext_vector_type(4)))  unsigned int u32x4;

static constexpr int Bc = 4;
static constexpr int Tc = 4096;
static constexpr int Dc = 1024;
static constexpr int Gc = 16;
static constexpr int Cc = 64;      // head dim
static constexpr int NV = 80;      // augmented V columns: 64 v + 1 ones + 15 pad
static constexpr int LP = 40;      // LDS tile row pitch in bf16 elems (80 B)

union FragU { v16bf v; u32x4 u[2]; };

// A-matrix fragment (16x32 bf16): lane<16 holds row M=lane%16, k = {0..7,16..23};
// lane>=16 same row, k = {8..15,24..31}. Caller pre-adds the 0/8 k-offset.
__device__ __forceinline__ v16bf load_a_frag(const __bf16* p) {
  FragU f;
  f.u[0] = *reinterpret_cast<const u32x4*>(p);        // k seg..seg+7
  f.u[1] = *reinterpret_cast<const u32x4*>(p + 16);   // k seg+16..seg+23
  return f.v;
}

// B-matrix fragment (32x16 bf16): lane holds column N=lane%16, k = {0..7,16..23}
// (lane<16) or {8..15,24..31} (lane>=16). Caller pre-adds the 0/8 k-offset.
__device__ __forceinline__ v16bf load_b_frag(const __bf16* p) {
  FragU f;
  f.u[0] = *reinterpret_cast<const u32x4*>(p);        // k .. k+7
  f.u[1] = *reinterpret_cast<const u32x4*>(p + 16);   // k+16 .. k+23
  return f.v;
}

__device__ __forceinline__ v8f wmma_bf16(v16bf a, v16bf b, v8f c) {
  // (neg_a, A, neg_b, B, c_mod, C, reuse_a, reuse_b)
  return __builtin_amdgcn_wmma_f32_16x16x32_bf16(false, a, false, b,
                                                 (short)0, c, false, false);
}

__device__ __forceinline__ float elu1(float x) {
  return (x > 0.0f) ? (x + 1.0f) : __expf(x);
}

// Raw 32-bit LDS offset from a generic shared pointer (its low 32 address bits).
__device__ __forceinline__ unsigned lds_off(const void* p) {
  return (unsigned)(unsigned long long)p;
}

// GLOBAL_LOAD_ASYNC_TO_LDS_B128: 32 lanes x 16 B, tracked by ASYNCcnt.
__device__ __forceinline__ void async_b128(unsigned ldsaddr, unsigned long long gaddr) {
  asm volatile("global_load_async_to_lds_b128 %0, %1, off"
               :: "v"(ldsaddr), "v"(gaddr) : "memory");
}

__device__ __forceinline__ void wait_async0() {
  asm volatile("s_wait_asynccnt 0" ::: "memory");
}

// ---------------------------------------------------------------------------
// f32 -> bf16 conversion (8 elems/thread: 32 B f32 in, b128 out)
// ---------------------------------------------------------------------------
__global__ void f32_to_bf16(const float* __restrict__ in,
                            __bf16* __restrict__ out, int n) {
  int i = (blockIdx.x * blockDim.x + threadIdx.x) * 8;
  const int stride = gridDim.x * blockDim.x * 8;
  for (; i + 7 < n; i += stride) {
    v8bf o;
#pragma unroll
    for (int e = 0; e < 8; ++e) o[e] = (__bf16)in[i + e];
    *reinterpret_cast<v8bf*>(out + i) = o;
  }
}

// ---------------------------------------------------------------------------
// Projection GEMM: Out[M,N] = act(A[M,K] @ W[N,K]^T + bias)
// M = B*T = 16384, N = K = D = 1024.
// Block = 256 threads (8 waves), tile 128x128; wave = 32x64 (2x4 WMMA tiles).
// A/B k-slabs (128x32 bf16) staged into LDS via async copy, double-buffered;
// fragments then read from LDS (pitch 80 B -> conflict-free ds_load_b128).
// ---------------------------------------------------------------------------
enum { MODE_QF = 0, MODE_KT = 1, MODE_VT = 2, MODE_OUT = 3 };

template <int MODE>
__global__ __launch_bounds__(256) void proj_gemm(
    const __bf16* __restrict__ A, const __bf16* __restrict__ W,
    const float* __restrict__ bias, const int* __restrict__ mask,
    void* __restrict__ outp) {
  __shared__ __bf16 shA[2][128 * LP];
  __shared__ __bf16 shB[2][128 * LP];

  const int K = Dc, N = Dc;
  const int lane = threadIdx.x & 31;
  const int wave = threadIdx.x >> 5;
  const int lm   = lane & 15;
  const bool hi  = lane >= 16;
  const int wm   = wave & 3;          // 4 waves along M
  const int wn   = wave >> 2;         // 2 waves along N
  const int rowblk = blockIdx.y * 128;
  const int colblk = blockIdx.x * 128;
  const int row0 = rowblk + wm * 32;
  const int col0 = colblk + wn * 64;

  v8f acc[2][4] = {};
  const int aseg = hi ? 8 : 0;
  const int bseg = hi ? 8 : 0;

  // Async-copy lane mapping: 32 lanes x 16B = 8 rows x 64B per instruction.
  // Each wave stages 16 rows of A and 16 rows of B per k-slab (4 asyncs).
  const int rl = lane >> 2;            // row within 8-row group
  const int cq = (lane & 3) * 8;       // k-element offset (0,8,16,24)
  const unsigned aoff[2] = { lds_off(&shA[0][0]), lds_off(&shA[1][0]) };
  const unsigned boff[2] = { lds_off(&shB[0][0]), lds_off(&shB[1][0]) };

  auto issue = [&](int buf, int kb) {
#pragma unroll
    for (int h = 0; h < 16; h += 8) {
      const int r = wave * 16 + rl + h;                    // tile row 0..127
      async_b128(aoff[buf] + (unsigned)(r * LP + cq) * 2,
                 (unsigned long long)(A + (size_t)(rowblk + r) * K + kb + cq));
      async_b128(boff[buf] + (unsigned)(r * LP + cq) * 2,
                 (unsigned long long)(W + (size_t)(colblk + r) * K + kb + cq));
    }
  };

  issue(0, 0);
  int cur = 0;
  for (int kb = 0; kb < K; kb += 32, cur ^= 1) {
    wait_async0();        // own current-buffer asyncs done
    __syncthreads();      // everyone's done -> buffer valid; prev readers done
    if (kb + 32 < K) issue(cur ^ 1, kb + 32);

    const __bf16* ap0 = &shA[cur][(wm * 32 + lm) * LP + aseg];
    const __bf16* ap1 = ap0 + 16 * LP;
    v16bf a0 = load_a_frag(ap0);
    v16bf a1 = load_a_frag(ap1);
#pragma unroll
    for (int j = 0; j < 4; ++j) {
      v16bf b = load_b_frag(&shB[cur][(wn * 64 + j * 16 + lm) * LP + bseg]);
      acc[0][j] = wmma_bf16(a0, b, acc[0][j]);
      acc[1][j] = wmma_bf16(a1, b, acc[1][j]);
    }
  }

  // Epilogue. C/D layout: lane holds column n=lm; VGPR r holds row r (+8 if hi).
  const int mb = hi ? 8 : 0;
#pragma unroll
  for (int i = 0; i < 2; ++i) {
#pragma unroll
    for (int j = 0; j < 4; ++j) {
      const int nn = col0 + j * 16 + lm;
      const float bv = bias[nn];
      if (MODE == MODE_KT || MODE == MODE_VT) {
        // Packed transposed store into [B, D, T]: per lane 8 consecutive t's.
        const int tb = row0 + i * 16 + mb;        // global flat row base
        const int b  = tb / Tc;                   // tile never straddles b
        const int t  = tb - b * Tc;
        v8bf pack;
#pragma unroll
        for (int r = 0; r < 8; ++r) {
          float v = acc[i][j][r] + bv;
          if (MODE == MODE_KT) {
            v = elu1(v);
            if (mask[b * Tc + t + r] == 0) v = 0.0f;
          }
          pack[r] = (__bf16)v;
        }
        __bf16* op = (__bf16*)outp;
        size_t off = ((size_t)b * Dc + nn) * Tc + t;
        *reinterpret_cast<v8bf*>(op + off) = pack;
      } else {
#pragma unroll
        for (int r = 0; r < 8; ++r) {
          const int m = row0 + i * 16 + mb + r;
          float v = acc[i][j][r] + bv;
          if (MODE == MODE_QF) {
            ((__bf16*)outp)[(size_t)m * N + nn] = (__bf16)elu1(v);
          } else {  // MODE_OUT: final fp32 output
            ((float*)outp)[(size_t)m * N + nn] = v;
          }
        }
      }
    }
  }
}

// ---------------------------------------------------------------------------
// Per-head KV' = [ K^T V | k_sum | 0 ]  (64 x 80), contraction over T=4096.
// kft/vpt are [B, D, T] bf16 (t contiguous). One block per (b,g); each of the
// 8 waves accumulates over a 512-wide T slice (4x5 WMMA tiles), then waves
// reduce through LDS ds_add_f32 atomics. Result stored transposed [BG, 80, 64].
// No cross-wave read redundancy -> direct global b128 fragment loads.
// ---------------------------------------------------------------------------
__global__ __launch_bounds__(256) void kv_head_gemm(
    const __bf16* __restrict__ kft, const __bf16* __restrict__ vpt,
    __bf16* __restrict__ kvt) {
  __shared__ float sacc[Cc * NV];
  const int bg   = blockIdx.x;
  const int lane = threadIdx.x & 31;
  const int wave = threadIdx.x >> 5;
  const int lm   = lane & 15;
  const bool hi  = lane >= 16;

  for (int i = threadIdx.x; i < Cc * NV; i += 256) sacc[i] = 0.0f;
  __syncthreads();

  v8f acc[4][5] = {};
  const int aseg = hi ? 8 : 0;
  const int bseg = hi ? 8 : 0;
  const size_t base = (size_t)bg * Cc * Tc;
  const int t0w = wave * (Tc / 8);

  const __bf16* aptr[4];
  const __bf16* bptr[4];
#pragma unroll
  for (int i = 0; i < 4; ++i)
    aptr[i] = kft + base + (size_t)(i * 16 + lm) * Tc + t0w + aseg;
#pragma unroll
  for (int j = 0; j < 4; ++j)
    bptr[j] = vpt + base + (size_t)(j * 16 + lm) * Tc + t0w + bseg;

  // Ones column (n == 64) lives in lanes with lm == 0 of the 5th n-tile.
  v16bf bone;
#pragma unroll
  for (int e = 0; e < 16; ++e) bone[e] = (lm == 0) ? (__bf16)1.0f : (__bf16)0.0f;

  for (int tk = 0; tk < Tc / 8; tk += 32) {
    v16bf af[4];
#pragma unroll
    for (int i = 0; i < 4; ++i) af[i] = load_a_frag(aptr[i] + tk);
#pragma unroll
    for (int j = 0; j < 5; ++j) {
      v16bf bfrag = (j < 4) ? load_b_frag(bptr[j] + tk) : bone;
#pragma unroll
      for (int i = 0; i < 4; ++i) acc[i][j] = wmma_bf16(af[i], bfrag, acc[i][j]);
    }
  }

  const int mb = hi ? 8 : 0;
#pragma unroll
  for (int i = 0; i < 4; ++i)
#pragma unroll
    for (int j = 0; j < 5; ++j)
#pragma unroll
      for (int r = 0; r < 8; ++r)
        atomicAdd(&sacc[(i * 16 + mb + r) * NV + j * 16 + lm], acc[i][j][r]);
  __syncthreads();

  // Store transposed: kvt[bg][n][c] = sacc[c][n] (bf16), n in [0,80), c in [0,64)
  for (int idx = threadIdx.x; idx < NV * Cc; idx += 256) {
    const int n = idx / Cc, c = idx % Cc;
    kvt[(size_t)bg * NV * Cc + idx] = (__bf16)sacc[c * NV + n];
  }
}

// ---------------------------------------------------------------------------
// o[b,t,g,:] = (qf[b,t,g,:] @ KV) / (qf[b,t,g,:] . k_sum)
// A = qf (row-major [B*T, D]); B = kvt [BG, 80, 64] (pre-transposed).
// Block: 128 t-rows x one head; 8 waves, each one 16-row tile x 5 n-tiles.
// Accumulator column 64 is the denominator.
// ---------------------------------------------------------------------------
__global__ __launch_bounds__(256) void attn_apply(
    const __bf16* __restrict__ qf, const __bf16* __restrict__ kvt,
    __bf16* __restrict__ obf) {
  __shared__ float den[128];
  const int lane = threadIdx.x & 31;
  const int wave = threadIdx.x >> 5;
  const int lm   = lane & 15;
  const bool hi  = lane >= 16;
  const int bg = blockIdx.y;
  const int b  = bg / Gc, g = bg % Gc;
  const int trow0 = blockIdx.x * 128 + wave * 16;
  const int aseg = hi ? 8 : 0;
  const int bseg = hi ? 8 : 0;

  v8f acc[5] = {};
  const __bf16* aptr  = qf + ((size_t)b * Tc + trow0 + lm) * Dc + g * Cc + aseg;
  const __bf16* bbase = kvt + (size_t)bg * NV * Cc + bseg;

#pragma unroll
  for (int kb = 0; kb < Cc; kb += 32) {
    v16bf a = load_a_frag(aptr + kb);
#pragma unroll
    for (int j = 0; j < 5; ++j) {
      v16bf bfrag = load_b_frag(bbase + (size_t)(j * 16 + lm) * Cc + kb);
      acc[j] = wmma_bf16(a, bfrag, acc[j]);
    }
  }

  const int mb = hi ? 8 : 0;
  if (lm == 0) {
#pragma unroll
    for (int r = 0; r < 8; ++r) den[wave * 16 + mb + r] = acc[4][r];
  }
  __syncthreads();

#pragma unroll
  for (int j = 0; j < 4; ++j) {
#pragma unroll
    for (int r = 0; r < 8; ++r) {
      const int m = mb + r;
      const float v = acc[j][r] / den[wave * 16 + m];
      obf[((size_t)b * Tc + trow0 + m) * Dc + g * Cc + j * 16 + lm] = (__bf16)v;
    }
  }
}

// ---------------------------------------------------------------------------
// Host launcher
// ---------------------------------------------------------------------------
extern "C" void kernel_launch(void* const* d_in, const int* in_sizes, int n_in,
                              void* d_out, int out_size, void* d_ws,
                              size_t ws_size, hipStream_t stream) {
  (void)in_sizes; (void)n_in; (void)out_size; (void)ws_size;
  const float* q    = (const float*)d_in[0];
  const float* k    = (const float*)d_in[1];
  const float* v    = (const float*)d_in[2];
  const int*   mask = (const int*)d_in[3];
  const float* Wq = (const float*)d_in[4];
  const float* bq = (const float*)d_in[5];
  const float* Wk = (const float*)d_in[6];
  const float* bk = (const float*)d_in[7];
  const float* Wv = (const float*)d_in[8];
  const float* bv = (const float*)d_in[9];
  const float* Wo = (const float*)d_in[10];
  const float* bo = (const float*)d_in[11];

  char* ws = (char*)d_ws;
  const size_t MB = 1ull << 20;
  __bf16* wqb = (__bf16*)(ws + 0 * MB);     // 2 MB each
  __bf16* wkb = (__bf16*)(ws + 2 * MB);
  __bf16* wvb = (__bf16*)(ws + 4 * MB);
  __bf16* wob = (__bf16*)(ws + 6 * MB);
  __bf16* xb  = (__bf16*)(ws + 8 * MB);     // 32 MB staging (reused as obf)
  __bf16* qf  = (__bf16*)(ws + 40 * MB);    // 32 MB
  __bf16* kft = (__bf16*)(ws + 72 * MB);    // 32 MB  [B, D, T]
  __bf16* vpt = (__bf16*)(ws + 104 * MB);   // 32 MB  [B, D, T]
  __bf16* kvt = (__bf16*)(ws + 136 * MB);   // 0.63 MB [BG, 80, 64]
  __bf16* obf = xb;                         // xb is free after the V GEMM

  const int DD = Dc * Dc;
  const int XN = Bc * Tc * Dc;
  const dim3 gg(Dc / 128, (Bc * Tc) / 128); // (8, 128)

  f32_to_bf16<<<256, 256, 0, stream>>>(Wq, wqb, DD);
  f32_to_bf16<<<256, 256, 0, stream>>>(Wk, wkb, DD);
  f32_to_bf16<<<256, 256, 0, stream>>>(Wv, wvb, DD);
  f32_to_bf16<<<256, 256, 0, stream>>>(Wo, wob, DD);

  // Q projection -> qf (ELU+1, row-major)
  f32_to_bf16<<<2048, 256, 0, stream>>>(q, xb, XN);
  proj_gemm<MODE_QF><<<gg, 256, 0, stream>>>(xb, wqb, bq, nullptr, qf);
  // K projection -> kft (ELU+1, masked, transposed [B,D,T])
  f32_to_bf16<<<2048, 256, 0, stream>>>(k, xb, XN);
  proj_gemm<MODE_KT><<<gg, 256, 0, stream>>>(xb, wkb, bk, mask, kft);
  // V projection -> vpt (transposed [B,D,T])
  f32_to_bf16<<<2048, 256, 0, stream>>>(v, xb, XN);
  proj_gemm<MODE_VT><<<gg, 256, 0, stream>>>(xb, wvb, bv, nullptr, vpt);

  // Per-head KV' accumulation (64 heads)
  kv_head_gemm<<<Bc * Gc, 256, 0, stream>>>(kft, vpt, kvt);
  // Apply: numerator / denominator -> obf
  attn_apply<<<dim3(Tc / 128, Bc * Gc), 256, 0, stream>>>(qf, kvt, obf);
  // Output projection -> d_out (fp32)
  proj_gemm<MODE_OUT><<<gg, 256, 0, stream>>>(obf, wob, bo, nullptr, d_out);
}
